// GCN_36120674959518
// MI455X (gfx1250) — compile-verified, matrix operand order checked by reading, not run on hardware
//
#include <hip/hip_runtime.h>
#include <hip/hip_bf16.h>

typedef __attribute__((ext_vector_type(16))) _Float16 v16h;
typedef __attribute__((ext_vector_type(8)))  _Float16 v8h;
typedef __attribute__((ext_vector_type(4)))  _Float16 v4h;
typedef __attribute__((ext_vector_type(8)))  float    v8f;

#define GRID_N 16
#define NODES  256
#define CH     256
#define BATCH  256

// deg[n] = 1 (self loop) + in-degree; in-edges of (i,j): left, up, up-left, up-right
__device__ __forceinline__ float node_deg(int i, int j) {
    float d = 1.0f;
    if (j > 0) d += 1.0f;
    if (i > 0) {
        d += 1.0f;
        if (j > 0)          d += 1.0f;
        if (j < GRID_N - 1) d += 1.0f;
    }
    return d;
}

// ISA 16-bit A/B 16x32 fragment: lanes 0-15 hold K {0..7,16..23}, lanes 16-31 hold K {8..15,24..31}
__device__ __forceinline__ v16h load_frag16x32(const _Float16* rowPtr, int kbase) {
    v8h lo = *(const v8h*)(rowPtr + kbase);
    v8h hi = *(const v8h*)(rowPtr + kbase + 16);
    v16h r;
#pragma unroll
    for (int q = 0; q < 8; ++q) { r[q] = lo[q]; r[q + 8] = hi[q]; }
    return r;
}

// Async global -> LDS direct copy (CDNA5, ASYNCcnt-tracked, no VGPR round trip).
__device__ __forceinline__ void async_ldg_b128(unsigned lds_byte_off, const void* gaddr) {
    asm volatile("global_load_async_to_lds_b128 %0, %1, off"
                 :: "v"(lds_byte_off), "v"(gaddr) : "memory");
}
#define WAIT_ASYNCCNT(n) asm volatile("s_wait_asynccnt %0" :: "i"(n) : "memory")

// [B, C, N] f32  ->  [B*N, C] f16, LDS-tiled transpose (conflict-free 32x33 tile)
__global__ void __launch_bounds__(256)
xpose_f32_to_f16(const float* __restrict__ x, _Float16* __restrict__ X0) {
    const int b   = blockIdx.x;
    const int tid = threadIdx.x;
    __shared__ float tile[32][33];
    const float* xb = x + (size_t)b * CH * NODES;
    _Float16*    ob = X0 + (size_t)b * NODES * CH;
    const int lc = tid >> 5;   // 0..7
    const int ln = tid & 31;   // 0..31
    for (int tc = 0; tc < 8; ++tc) {
        for (int tn = 0; tn < 8; ++tn) {
#pragma unroll
            for (int q = 0; q < 4; ++q)
                tile[lc + q * 8][ln] =
                    xb[(size_t)(tc * 32 + lc + q * 8) * NODES + tn * 32 + ln];
            __syncthreads();
#pragma unroll
            for (int q = 0; q < 4; ++q) {
                int n = tn * 32 + lc + q * 8;
                ob[(size_t)n * CH + tc * 32 + ln] = (_Float16)tile[ln][lc + q * 8];
            }
            __syncthreads();
        }
    }
}

// One workgroup: graph b (256 nodes) x 64 output channels.
// Phase 1: H = X[b] * W_slice^T via v_wmma_f32_16x16x32_f16, A staged by
//          double-buffered global_load_async_to_lds_b128.
// Phase 2: normalized sparse aggregation + bias (+ReLU) in f32, fused via LDS.
__global__ void __launch_bounds__(256)
gcn_layer(const _Float16* __restrict__ Xin,
          const float*    __restrict__ Wt,     // [256 out][256 in] f32
          const float*    __restrict__ bias,   // [256]
          _Float16*       __restrict__ Xout,   // [B*N, 256] f16 (layers 1,2)
          float*          __restrict__ Yout,   // [B*N, 256] f32 (final layer) or null
          int relu)
{
    // LDS map (64 KB): [0,16K) bufA0 | [16K,32K) bufA1 | [32K,64K) Wall[64][256] f16
    // Aggregation phase aliases the whole region as Hs[256][64] f32.
    __shared__ __align__(16) unsigned char smemRaw[65536];
    _Float16 (*Wall)[CH] = reinterpret_cast<_Float16(*)[CH]>(smemRaw + 32768);
    float    (*Hs)[64]   = reinterpret_cast<float(*)[64]>(smemRaw);

    const int tid   = threadIdx.x;
    const int lane  = tid & 31;
    const int wave  = tid >> 5;            // 8 waves; wave owns M-tiles {2w, 2w+1}
    const int b     = blockIdx.y;
    const int obase = blockIdx.x * 64;

    const _Float16* Xb = Xin + ((size_t)b << 16);   // b * 256 * 256
    const unsigned smemBase = (unsigned)(uintptr_t)(void*)smemRaw;

    // ---- stage the whole 64x256 W slice once (f32 -> f16), coalesced ----
#pragma unroll
    for (int q = 0; q < 16; ++q) {
        int e = tid + 256 * q;             // float4 index into [64][256]
        int o = e >> 6, kv = (e & 63) * 4;
        const float4 w4 = *(const float4*)(Wt + (size_t)(obase + o) * CH + kv);
        v4h h4; h4[0] = (_Float16)w4.x; h4[1] = (_Float16)w4.y;
                h4[2] = (_Float16)w4.z; h4[3] = (_Float16)w4.w;
        *(v4h*)(&Wall[o][kv]) = h4;
    }

    // ---- async-stage helper: chunk kk (256 rows x 32 halfs) into buffer bsel ----
    auto asyncStage = [&](int kk, int bsel) {
        const _Float16* gbase = Xb + kk * 32;
        const unsigned  dbase = smemBase + (unsigned)bsel * 16384u;
#pragma unroll
        for (int q = 0; q < 4; ++q) {
            int e = tid + 256 * q;         // b128 index: lane-consecutive, coalesced
            int n = e >> 2, v = e & 3;
            async_ldg_b128(dbase + (unsigned)e * 16u, gbase + (size_t)n * CH + v * 8);
        }
    };

    const v8f vzero = {0.f, 0.f, 0.f, 0.f, 0.f, 0.f, 0.f, 0.f};
    v8f acc[2][4];
#pragma unroll
    for (int mt = 0; mt < 2; ++mt)
#pragma unroll
        for (int nt = 0; nt < 4; ++nt) acc[mt][nt] = vzero;

    const int kbase = (lane & 16) ? 8 : 0;

    asyncStage(0, 0);                       // preload K-chunk 0

#pragma unroll
    for (int kk = 0; kk < 8; ++kk) {
        if (kk < 7) {
            asyncStage(kk + 1, (kk + 1) & 1);   // overlap next copy with this chunk
            WAIT_ASYNCCNT(4);               // in-order: oldest 4 (chunk kk) landed
        } else {
            WAIT_ASYNCCNT(0);
        }
        __syncthreads();                    // all waves' copies (and W at kk=0) visible

        const _Float16* As = reinterpret_cast<const _Float16*>(smemRaw + (kk & 1) * 16384);
        v16h afrag[2], bfrag[4];
#pragma unroll
        for (int mt = 0; mt < 2; ++mt)
            afrag[mt] = load_frag16x32(As + ((wave * 2 + mt) * 16 + (lane & 15)) * 32, kbase);
#pragma unroll
        for (int nt = 0; nt < 4; ++nt)
            bfrag[nt] = load_frag16x32(&Wall[nt * 16 + (lane & 15)][kk * 32], kbase);

#pragma unroll
        for (int mt = 0; mt < 2; ++mt)
#pragma unroll
            for (int nt = 0; nt < 4; ++nt)
                acc[mt][nt] = __builtin_amdgcn_wmma_f32_16x16x32_f16(
                    false, afrag[mt], false, bfrag[nt],
                    (short)0, acc[mt][nt], false, false);
        __syncthreads();                    // readers done before buffer reuse
    }

    // ---- spill accumulators to Hs[256][64] (C/D layout: VGPR r -> M=r / r+8) ----
    const int roff = (lane & 16) ? 8 : 0;
    const int col  = lane & 15;
#pragma unroll
    for (int mt = 0; mt < 2; ++mt) {
        int rowb = (wave * 2 + mt) * 16 + roff;
#pragma unroll
        for (int nt = 0; nt < 4; ++nt)
#pragma unroll
            for (int r = 0; r < 8; ++r)
                Hs[rowb + r][nt * 16 + col] = acc[mt][nt][r];
    }
    __syncthreads();

    // ---- fused aggregation: channel-per-lane (bank-conflict-free), node loop ----
    const int c   = tid & 63;
    const int grp = tid >> 6;
    const float bv = bias[obase + c];
    for (int n0 = 0; n0 < 64; ++n0) {
        int n = grp * 64 + n0;             // wave-uniform -> uniform branches
        int i = n >> 4, j = n & 15;
        float rn = rsqrtf(node_deg(i, j));
        float a  = rn * rn * Hs[n][c];     // self loop: 1/deg
        if (j > 0)              a += rn * rsqrtf(node_deg(i, j - 1)) * Hs[n - 1][c];
        if (i > 0) {
                                a += rn * rsqrtf(node_deg(i - 1, j)) * Hs[n - 16][c];
            if (j > 0)          a += rn * rsqrtf(node_deg(i - 1, j - 1)) * Hs[n - 17][c];
            if (j < GRID_N - 1) a += rn * rsqrtf(node_deg(i - 1, j + 1)) * Hs[n - 15][c];
        }
        a += bv;
        if (relu) a = fmaxf(a, 0.0f);
        size_t idx = (((size_t)b * NODES + n) << 8) + obase + c;   // [B*N, Cout] (raw reshape)
        if (Yout) Yout[idx] = a;
        else      Xout[idx] = (_Float16)a;
    }
}

extern "C" void kernel_launch(void* const* d_in, const int* in_sizes, int n_in,
                              void* d_out, int out_size, void* d_ws, size_t ws_size,
                              hipStream_t stream) {
    const float* x  = (const float*)d_in[0];
    const float* W1 = (const float*)d_in[1];
    const float* b1 = (const float*)d_in[2];
    const float* W2 = (const float*)d_in[3];
    const float* b2 = (const float*)d_in[4];
    const float* W3 = (const float*)d_in[5];
    const float* b3 = (const float*)d_in[6];
    float* out = (float*)d_out;

    _Float16* X0 = (_Float16*)d_ws;                       // 32 MB
    _Float16* X1 = X0 + (size_t)BATCH * NODES * CH;       // 32 MB

    xpose_f32_to_f16<<<dim3(BATCH), dim3(256), 0, stream>>>(x, X0);

    dim3 g(4, BATCH);   // 4 channel slices x 256 graphs
    gcn_layer<<<g, 256, 0, stream>>>(X0, W1, b1, X1, nullptr, 1);
    gcn_layer<<<g, 256, 0, stream>>>(X1, W2, b2, X0, nullptr, 1);
    gcn_layer<<<g, 256, 0, stream>>>(X0, W3, b3, nullptr, out, 0);
}
